// GraphSemanticExtractor_2946347565060
// MI455X (gfx1250) — compile-verified, matrix operand order checked by reading, not run on hardware
//
#include <hip/hip_runtime.h>
#include <hip/hip_bf16.h>

typedef __attribute__((ext_vector_type(16))) __bf16 v16bf;
typedef __attribute__((ext_vector_type(8)))  float  v8f;

#define B_     2
#define S_     1024
#define H_     1024
#define HEADS_ 4
#define KNN_   8
#define SEM_   512
#define NL_    2
#define N_     (B_ * S_)   // 2048 nodes

__device__ __forceinline__ unsigned short f2bf(float f) {
  union { float f; unsigned u; } x; x.f = f;
  unsigned u = x.u;
  unsigned rnd = ((u >> 16) & 1u) + 0x7FFFu;           // round-to-nearest-even
  return (unsigned short)((u + rnd) >> 16);
}

// ---------------------------------------------------------------------------
// fp32 -> bf16 conversion (grid-stride)
// ---------------------------------------------------------------------------
__global__ void f32_to_bf16_kernel(const float* __restrict__ src,
                                   unsigned short* __restrict__ dst, size_t n) {
  size_t i = (size_t)blockIdx.x * blockDim.x + threadIdx.x;
  size_t stride = (size_t)gridDim.x * blockDim.x;
  for (; i < n; i += stride) dst[i] = f2bf(src[i]);
}

// ---------------------------------------------------------------------------
// NT GEMM via v_wmma_f32_16x16x32_bf16, 2x2 register blocking:
//   C[M,N] = A[M,K] @ B[N,K]^T,  A/B bf16 row-major (ld = K), C row-major (ld = N)
// One wave computes a 32x32 C region (4 WMMA tiles). Per K-step: 4 fragment
// loads feed 4 WMMAs (2 b128 loads per WMMA). Guard is wave-uniform so EXEC
// stays all-ones across the WMMAs.
// A frag: lane<16 holds row (lane), K=k0..k0+15; lane>=16 same rows K=k0+16..31.
// B frag (column n of B == row n of Bmat): identical addressing on Bmat.
// C frag: VGPR i -> row i (+8 for upper lanes), col = lane&15.
// ---------------------------------------------------------------------------
__global__ void gemm_bf16_nt(const unsigned short* __restrict__ A,
                             const unsigned short* __restrict__ Bm,
                             float* __restrict__ Cf, unsigned short* __restrict__ Cb,
                             int M, int N, int Kd) {
  int gw   = (int)((blockIdx.x * blockDim.x + threadIdx.x) >> 5);
  int lane = (int)(threadIdx.x & 31);
  int tilesN = N >> 5;                              // 32-wide wave tiles
  int tM = gw / tilesN;
  int tN = gw - tM * tilesN;
  if (tM >= (M >> 5)) return;                       // wave-uniform
  int r = lane & 15, hsel = lane >> 4;
  const unsigned short* a0 = A  + (size_t)(tM * 32 + r) * Kd + hsel * 16;
  const unsigned short* a1 = a0 + (size_t)16 * Kd;
  const unsigned short* b0 = Bm + (size_t)(tN * 32 + r) * Kd + hsel * 16;
  const unsigned short* b1 = b0 + (size_t)16 * Kd;
  v8f acc00 = {}, acc01 = {}, acc10 = {}, acc11 = {};
  for (int k0 = 0; k0 < Kd; k0 += 32) {
    v16bf av0 = *(const v16bf*)(a0 + k0);
    v16bf av1 = *(const v16bf*)(a1 + k0);
    v16bf bv0 = *(const v16bf*)(b0 + k0);
    v16bf bv1 = *(const v16bf*)(b1 + k0);
    __builtin_prefetch(a0 + k0 + 128, 0, 1);        // global_prefetch_b8
    __builtin_prefetch(a1 + k0 + 128, 0, 1);
    __builtin_prefetch(b0 + k0 + 128, 0, 1);
    __builtin_prefetch(b1 + k0 + 128, 0, 1);
    acc00 = __builtin_amdgcn_wmma_f32_16x16x32_bf16(
        false, av0, false, bv0, (short)0, acc00, false, false);
    acc01 = __builtin_amdgcn_wmma_f32_16x16x32_bf16(
        false, av0, false, bv1, (short)0, acc01, false, false);
    acc10 = __builtin_amdgcn_wmma_f32_16x16x32_bf16(
        false, av1, false, bv0, (short)0, acc10, false, false);
    acc11 = __builtin_amdgcn_wmma_f32_16x16x32_bf16(
        false, av1, false, bv1, (short)0, acc11, false, false);
  }
  int col0 = tN * 32 + r;
  auto store_tile = [&](v8f acc, int rowBase, int col) {
#pragma unroll
    for (int i = 0; i < 8; ++i) {
      int row = rowBase + i + hsel * 8;
      size_t off = (size_t)row * N + col;
      float v = acc[i];
      if (Cf) Cf[off] = v;
      if (Cb) Cb[off] = f2bf(v);
    }
  };
  store_tile(acc00, tM * 32,      col0);
  store_tile(acc01, tM * 32,      col0 + 16);
  store_tile(acc10, tM * 32 + 16, col0);
  store_tile(acc11, tM * 32 + 16, col0 + 16);
}

// ---------------------------------------------------------------------------
// Top-8 per score row -> edge list (dst index + softmaxed weight, self zeroed)
// One wave per row; per-lane insertion top-8, lane0 merges via LDS.
// ---------------------------------------------------------------------------
__global__ void topk_edges_kernel(const float* __restrict__ scores,
                                  int* __restrict__ dstI, float* __restrict__ ewB) {
  __shared__ float sv[8 * 32 * KNN_];
  __shared__ int   si[8 * 32 * KNN_];
  int wl = (int)(threadIdx.x >> 5);
  int lane = (int)(threadIdx.x & 31);
  int row = blockIdx.x * 8 + wl;                 // node id, 0..N_-1
  int b = row >> 10, s = row & (S_ - 1);
  const float* sr = scores + (size_t)row * S_;
  float v[KNN_]; int id[KNN_];
#pragma unroll
  for (int i = 0; i < KNN_; ++i) { v[i] = -3.0e38f; id[i] = 0; }
  for (int t = lane; t < S_; t += 32) {
    float x = sr[t];
    if (x > v[KNN_ - 1]) {
      int j = KNN_ - 1;
      while (j > 0 && x > v[j - 1]) { v[j] = v[j - 1]; id[j] = id[j - 1]; --j; }
      v[j] = x; id[j] = t;
    }
  }
  float* svw = sv + wl * 32 * KNN_;
  int*   siw = si + wl * 32 * KNN_;
#pragma unroll
  for (int i = 0; i < KNN_; ++i) { svw[lane * KNN_ + i] = v[i]; siw[lane * KNN_ + i] = id[i]; }
  __syncthreads();
  if (lane == 0) {
    float bv[KNN_]; int bi[KNN_];
#pragma unroll
    for (int i = 0; i < KNN_; ++i) { bv[i] = -3.0e38f; bi[i] = 0; }
    for (int c = 0; c < 32 * KNN_; ++c) {
      float x = svw[c];
      if (x > bv[KNN_ - 1]) {
        int j = KNN_ - 1;
        while (j > 0 && x > bv[j - 1]) { bv[j] = bv[j - 1]; bi[j] = bi[j - 1]; --j; }
        bv[j] = x; bi[j] = siw[c];
      }
    }
    float m = bv[0], sum = 0.f, e[KNN_];
#pragma unroll
    for (int i = 0; i < KNN_; ++i) { e[i] = __expf(bv[i] - m); sum += e[i]; }
    float inv = 1.f / (sum + 1e-8f * __expf(-m));
#pragma unroll
    for (int i = 0; i < KNN_; ++i) {
      int ti = bi[i];
      float w = e[i] * inv;
      dstI[row * KNN_ + i] = b * S_ + ti;
      ewB [row * KNN_ + i] = (ti != s) ? fmaxf(w, 1e-8f) : 0.f;
    }
  }
}

// ---------------------------------------------------------------------------
// Per-(node, head) attention logits: e_src = <h, a_src>, e_dst = <h, a_dst>
// ---------------------------------------------------------------------------
__global__ void edge_score_kernel(const float* __restrict__ h, const float* __restrict__ att,
                                  float* __restrict__ esrc, float* __restrict__ edst) {
  int n = blockIdx.x, hd = blockIdx.y;
  const float* hv = h + ((size_t)n * HEADS_ + hd) * H_;
  const float* as = att + (size_t)hd * 2 * H_;
  const float* ad = as + H_;
  float s0 = 0.f, s1 = 0.f;
  for (int t = threadIdx.x; t < H_; t += blockDim.x) {
    float x = hv[t]; s0 += x * as[t]; s1 += x * ad[t];
  }
  __shared__ float r0[128], r1[128];
  int tid = threadIdx.x;
  r0[tid] = s0; r1[tid] = s1; __syncthreads();
  for (int st = 64; st > 0; st >>= 1) {
    if (tid < st) { r0[tid] += r0[tid + st]; r1[tid] += r1[tid + st]; }
    __syncthreads();
  }
  if (tid == 0) { esrc[n * HEADS_ + hd] = r0[0]; edst[n * HEADS_ + hd] = r1[0]; }
}

// ---------------------------------------------------------------------------
// Edge scatter: out[dst] += wgt * h[src], attn[dst] += wgt   (segment_sum)
// ---------------------------------------------------------------------------
__global__ void scatter_kernel(const float* __restrict__ h, const int* __restrict__ dstI,
                               const float* __restrict__ ewB, const float* __restrict__ esrc,
                               const float* __restrict__ edst, float* __restrict__ outb,
                               float* __restrict__ attn) {
  int e = blockIdx.x, hd = blockIdx.y;
  float w = ewB[e];
  if (w == 0.f) return;
  int sN = e >> 3;        // src node = edge / KNN
  int dN = dstI[e];
  float x = esrc[sN * HEADS_ + hd] + edst[dN * HEADS_ + hd];
  float lr = (x > 0.f) ? x : 0.2f * x;             // leaky_relu(0.2)
  float wgt = w * __expf(lr);
  const float* hp = h + ((size_t)sN * HEADS_ + hd) * H_;
  float* op = outb + ((size_t)dN * HEADS_ + hd) * H_;
  for (int t = threadIdx.x; t < H_; t += blockDim.x)
    atomicAdd(&op[t], wgt * hp[t]);
  if (threadIdx.x == 0) atomicAdd(&attn[dN * HEADS_ + hd], wgt);
}

// ---------------------------------------------------------------------------
// Normalize by attn, mean over heads, relu
// ---------------------------------------------------------------------------
__global__ void finalize_kernel(const float* __restrict__ outb, const float* __restrict__ attn,
                                float* __restrict__ x) {
  int idx = blockIdx.x * blockDim.x + threadIdx.x;   // N_*H_
  int n = idx >> 10, t = idx & (H_ - 1);
  float acc = 0.f;
#pragma unroll
  for (int hd = 0; hd < HEADS_; ++hd)
    acc += outb[((size_t)n * HEADS_ + hd) * H_ + t] / (attn[n * HEADS_ + hd] + 1e-8f);
  acc *= (1.f / HEADS_);
  x[idx] = acc > 0.f ? acc : 0.f;
}

// ---------------------------------------------------------------------------
// Pooling: per-node score, per-batch softmax over S, weighted sum
// ---------------------------------------------------------------------------
__global__ void pool_score_kernel(const float* __restrict__ x, const float* __restrict__ wp,
                                  float* __restrict__ ps) {
  int n = blockIdx.x;
  float s = 0.f;
  for (int t = threadIdx.x; t < H_; t += blockDim.x) s += x[(size_t)n * H_ + t] * wp[t];
  __shared__ float r[128];
  r[threadIdx.x] = s; __syncthreads();
  for (int st = 64; st > 0; st >>= 1) {
    if (threadIdx.x < st) r[threadIdx.x] += r[threadIdx.x + st];
    __syncthreads();
  }
  if (threadIdx.x == 0) ps[n] = r[0];
}

__global__ void softmax_kernel(const float* __restrict__ ps, float* __restrict__ alpha) {
  int b = blockIdx.x;
  const float* p = ps + b * S_;
  __shared__ float r[256];
  int tid = threadIdx.x;
  float m = -3.0e38f;
  for (int t = tid; t < S_; t += 256) m = fmaxf(m, p[t]);
  r[tid] = m; __syncthreads();
  for (int st = 128; st > 0; st >>= 1) {
    if (tid < st) r[tid] = fmaxf(r[tid], r[tid + st]);
    __syncthreads();
  }
  m = r[0]; __syncthreads();
  float s = 0.f;
  for (int t = tid; t < S_; t += 256) s += __expf(p[t] - m);
  r[tid] = s; __syncthreads();
  for (int st = 128; st > 0; st >>= 1) {
    if (tid < st) r[tid] += r[tid + st];
    __syncthreads();
  }
  float inv = 1.f / r[0];
  for (int t = tid; t < S_; t += 256) alpha[b * S_ + t] = __expf(p[t] - m) * inv;
}

__global__ void pooled_kernel(const float* __restrict__ x, const float* __restrict__ alpha,
                              float* __restrict__ pooled) {
  int b = blockIdx.y;
  int d = blockIdx.x * blockDim.x + threadIdx.x;     // H_
  float acc = 0.f;
  for (int s = 0; s < S_; ++s)
    acc += alpha[b * S_ + s] * x[((size_t)(b * S_ + s)) * H_ + d];
  pooled[b * H_ + d] = acc;
}

__global__ void mlp1_kernel(const float* __restrict__ pooled, const float* __restrict__ w1,
                            const float* __restrict__ b1, float* __restrict__ hidden) {
  int idx = blockIdx.x * blockDim.x + threadIdx.x;   // B_*(H_/2)
  int b = idx >> 9, j = idx & 511;
  const float* p  = pooled + b * H_;
  const float* wr = w1 + (size_t)j * H_;
  float acc = b1[j];
  for (int k = 0; k < H_; ++k) acc += p[k] * wr[k];
  hidden[idx] = acc > 0.f ? acc : 0.f;
}

__global__ void mlp2_kernel(const float* __restrict__ hidden, const float* __restrict__ w2,
                            const float* __restrict__ b2, float* __restrict__ out) {
  int idx = blockIdx.x * blockDim.x + threadIdx.x;   // B_*SEM_
  int b = idx >> 9, j = idx & 511;
  const float* hp = hidden + b * (H_ / 2);
  const float* wr = w2 + (size_t)j * (H_ / 2);
  float acc = b2[j];
  for (int k = 0; k < H_ / 2; ++k) acc += hp[k] * wr[k];
  out[idx] = acc;
}

// ---------------------------------------------------------------------------
extern "C" void kernel_launch(void* const* d_in, const int* in_sizes, int n_in,
                              void* d_out, int out_size, void* d_ws, size_t ws_size,
                              hipStream_t stream) {
  const float* hs      = (const float*)d_in[0];
  const float* phi_w   = (const float*)d_in[1];
  const float* psi_w   = (const float*)d_in[2];
  const float* lin_w   = (const float*)d_in[3];
  const float* gat_att = (const float*)d_in[4];
  const float* wp      = (const float*)d_in[5];
  const float* w1      = (const float*)d_in[6];
  const float* b1      = (const float*)d_in[7];
  const float* w2      = (const float*)d_in[8];
  const float* b2      = (const float*)d_in[9];
  float* out = (float*)d_out;

  char* ws = (char*)d_ws;
  size_t off = 0;
  auto take = [&](size_t bytes) -> char* {
    char* p = ws + off;
    off = (off + bytes + 255) & ~(size_t)255;
    return p;
  };
  unsigned short* a_bf = (unsigned short*)take((size_t)N_ * H_ * 2);
  unsigned short* w_bf = (unsigned short*)take((size_t)HEADS_ * H_ * H_ * 2);
  unsigned short* phiH = (unsigned short*)take((size_t)N_ * H_ * 2);
  unsigned short* psiH = (unsigned short*)take((size_t)N_ * H_ * 2);
  float* hbuf   = (float*)take((size_t)N_ * HEADS_ * H_ * 4);  // 32 MB; scores alias (8 MB)
  float* scores = hbuf;                                        // dead after top-k
  int*   dstI   = (int*)  take((size_t)N_ * KNN_ * 4);
  float* ewB    = (float*)take((size_t)N_ * KNN_ * 4);
  float* esrc   = (float*)take((size_t)N_ * HEADS_ * 4);
  float* edst   = (float*)take((size_t)N_ * HEADS_ * 4);
  float* outb   = (float*)take((size_t)N_ * HEADS_ * H_ * 4);
  float* attn   = (float*)take((size_t)N_ * HEADS_ * 4);
  float* xA     = (float*)take((size_t)N_ * H_ * 4);
  float* xB     = (float*)take((size_t)N_ * H_ * 4);
  float* ps     = (float*)take((size_t)N_ * 4);
  float* alpha  = (float*)take((size_t)N_ * 4);
  float* pooled = (float*)take((size_t)B_ * H_ * 4);
  float* hidden = (float*)take((size_t)B_ * (H_ / 2) * 4);

  // ---- edge construction -------------------------------------------------
  f32_to_bf16_kernel<<<2048, 256, 0, stream>>>(hs, a_bf, (size_t)N_ * H_);
  f32_to_bf16_kernel<<<2048, 256, 0, stream>>>(phi_w, w_bf, (size_t)H_ * H_);
  // phi_h (bf16): M=2048,N=1024 -> (64*32)=2048 wave-tiles / 8 waves per block
  gemm_bf16_nt<<<256, 256, 0, stream>>>(a_bf, w_bf, nullptr, phiH, N_, H_, H_);
  f32_to_bf16_kernel<<<2048, 256, 0, stream>>>(psi_w, w_bf, (size_t)H_ * H_);
  gemm_bf16_nt<<<256, 256, 0, stream>>>(a_bf, w_bf, nullptr, psiH, N_, H_, H_);
  // per-batch scores = phi_h @ psi_h^T : M=N=1024 -> (32*32)=1024 wave-tiles
  for (int b = 0; b < B_; ++b)
    gemm_bf16_nt<<<128, 256, 0, stream>>>(phiH + (size_t)b * S_ * H_,
                                          psiH + (size_t)b * S_ * H_,
                                          scores + (size_t)b * S_ * S_, nullptr,
                                          S_, S_, H_);
  topk_edges_kernel<<<N_ / 8, 256, 0, stream>>>(scores, dstI, ewB);

  // ---- GAT layers --------------------------------------------------------
  const float* xin = hs;
  float* xcur[2] = {xA, xB};
  for (int l = 0; l < NL_; ++l) {
    f32_to_bf16_kernel<<<2048, 256, 0, stream>>>(xin, a_bf, (size_t)N_ * H_);
    f32_to_bf16_kernel<<<4096, 256, 0, stream>>>(lin_w + (size_t)l * HEADS_ * H_ * H_,
                                                 w_bf, (size_t)HEADS_ * H_ * H_);
    // h = x @ lin_w^T : M=2048, N=4096 -> (64*128)=8192 wave-tiles
    gemm_bf16_nt<<<1024, 256, 0, stream>>>(a_bf, w_bf, hbuf, nullptr, N_, HEADS_ * H_, H_);
    dim3 gs(N_, HEADS_);
    edge_score_kernel<<<gs, 128, 0, stream>>>(hbuf, gat_att + (size_t)l * HEADS_ * 2 * H_,
                                              esrc, edst);
    hipMemsetAsync(outb, 0, (size_t)N_ * HEADS_ * H_ * 4, stream);
    hipMemsetAsync(attn, 0, (size_t)N_ * HEADS_ * 4, stream);
    dim3 ge(N_ * KNN_, HEADS_);
    scatter_kernel<<<ge, 256, 0, stream>>>(hbuf, dstI, ewB, esrc, edst, outb, attn);
    finalize_kernel<<<(N_ * H_) / 256, 256, 0, stream>>>(outb, attn, xcur[l]);
    xin = xcur[l];
  }

  // ---- attention pooling + MLP ------------------------------------------
  pool_score_kernel<<<N_, 128, 0, stream>>>(xin, wp, ps);
  softmax_kernel<<<B_, 256, 0, stream>>>(ps, alpha);
  dim3 gp(H_ / 256, B_);
  pooled_kernel<<<gp, 256, 0, stream>>>(xin, alpha, pooled);
  mlp1_kernel<<<(B_ * (H_ / 2)) / 256, 256, 0, stream>>>(pooled, w1, b1, hidden);
  mlp2_kernel<<<(B_ * SEM_) / 256, 256, 0, stream>>>(hidden, w2, b2, out);
}
